// RecursiveMamba130M_43250320671003
// MI455X (gfx1250) — matrix operand order repaired and checked by reading, not compiled, not run on hardware
//
#include <hip/hip_runtime.h>
#include <hip/hip_bf16.h>
#include <cmath>

// ---------------- model constants (match reference) ----------------
constexpr int   C_V   = 50280;
constexpr int   C_D   = 768;
constexpr int   C_DI  = 1536;
constexpr int   C_N   = 16;     // SSM states
constexpr int   C_RK  = 48;     // dt rank
constexpr int   C_LR  = 8;      // LoRA rank
constexpr float C_LS  = 2.0f;   // LoRA scale
constexpr int   C_NB  = 6;
constexpr int   C_NT  = 18;
constexpr int   C_L   = 256;
constexpr float C_THR = 0.85f;
constexpr float C_EPS = 1e-5f;
constexpr int   C_DBL = C_RK + 2 * C_N;   // 80

typedef __bf16 v16bf __attribute__((ext_vector_type(16)));
typedef float  v8f   __attribute__((ext_vector_type(8)));

enum { EPI_NONE = 0, EPI_BIAS_SOFTPLUS = 1 };

// ---------------- WMMA GEMM:  C[M,N] (+)= alpha * act[M,K] * W[N,K]^T ----------------
// One wave per (MT x 1) super-tile of 16x16 tiles: B (weight) fragment is loaded once
// per 32-K step and reused in registers for MT WMMAs (weights dominate traffic).
// All edge handling is branchless (clamped addresses + value selects) so EXEC stays
// all-ones around the WMMAs and loads stay vectorized.
template <int MT>
__global__ __launch_bounds__(256) void k_gemm(
    const float* __restrict__ act, int lda,
    const float* __restrict__ W,          // row-major (N, K)
    const float* __restrict__ bias,       // used by EPI_BIAS_SOFTPLUS
    float* __restrict__ C, int ldc,
    int M, int N, int K, float alpha, int accum, int epi)
{
  const int lane = threadIdx.x & 31;
  const int wid  = (blockIdx.x * blockDim.x + threadIdx.x) >> 5;
  const int nw   = (gridDim.x * blockDim.x) >> 5;
  const int tm = M / (16 * MT);
  const int tn = (N + 15) >> 4;
  const int hi  = lane >> 4;       // lane half selects K sub-block
  const int l15 = lane & 15;

  for (int t = wid; t < tm * tn; t += nw) {
    const int m0 = (t / tn) * (16 * MT);
    const int n0 = (t % tn) << 4;
    const int ncol = n0 + l15;
    const bool nv = (ncol < N);
    const float* arow0 = act + (size_t)(m0 + l15) * lda;
    const float* wrow  = W + (size_t)(nv ? ncol : 0) * K;
    v8f acc[MT];
#pragma unroll
    for (int mt = 0; mt < MT; ++mt) acc[mt] = {};

    for (int kk = 0; kk < K; kk += 32) {
      const int ka = kk + hi * 16;   // A: lanes 0-15 -> K 0-15,      lanes 16-31 -> 16-31
      const int kb = kk + hi * 16;   // B: lanes 0-15 -> K 0-15,      lanes 16-31 -> 16-31
      v16bf b;
      if (kk + 32 <= K) {            // fast path: fully in-bounds, vectorizable loads
        if (kk + 64 <= K) __builtin_prefetch(wrow + kk + 32, 0, 3);
#pragma unroll
        for (int e = 0; e < 16; ++e) b[e] = (__bf16)wrow[kb + e];
#pragma unroll
        for (int mt = 0; mt < MT; ++mt) {
          const float* arow = arow0 + (size_t)mt * 16 * lda;
          v16bf a;
#pragma unroll
          for (int e = 0; e < 8; ++e) {
            a[e]     = (__bf16)arow[ka + e];
            a[e + 8] = (__bf16)arow[ka + 8 + e];
          }
          acc[mt] = __builtin_amdgcn_wmma_f32_16x16x32_bf16(false, a, false, b,
                                                            (short)0, acc[mt], false, false);
        }
      } else {                       // K tail (K=8, 48): branchless clamp + mask
#pragma unroll
        for (int e = 0; e < 16; ++e) {
          const int k1 = kb + e;
          const float bv = wrow[(k1 < K) ? k1 : (K - 1)];
          b[e] = (__bf16)((k1 < K) ? bv : 0.f);
        }
#pragma unroll
        for (int mt = 0; mt < MT; ++mt) {
          const float* arow = arow0 + (size_t)mt * 16 * lda;
          v16bf a;
#pragma unroll
          for (int e = 0; e < 8; ++e) {
            const int k1 = ka + e, k2 = ka + 8 + e;
            const float a1 = arow[(k1 < K) ? k1 : (K - 1)];
            const float a2 = arow[(k2 < K) ? k2 : (K - 1)];
            a[e]     = (__bf16)((k1 < K) ? a1 : 0.f);
            a[e + 8] = (__bf16)((k2 < K) ? a2 : 0.f);
          }
          acc[mt] = __builtin_amdgcn_wmma_f32_16x16x32_bf16(false, a, false, b,
                                                            (short)0, acc[mt], false, false);
        }
      }
    }
    if (nv) {
#pragma unroll
      for (int mt = 0; mt < MT; ++mt) {
        const int r0 = m0 + mt * 16 + hi * 8;   // C/D layout: row = r + 8*(lane>=16)
#pragma unroll
        for (int r = 0; r < 8; ++r) {
          float v = alpha * acc[mt][r];
          float* cp = C + (size_t)(r0 + r) * ldc + ncol;
          if (accum) v += *cp;
          if (epi == EPI_BIAS_SOFTPLUS) {
            v += bias[ncol];
            v = (v > 20.f) ? v : log1pf(__expf(v));
          }
          *cp = v;
        }
      }
    }
  }
}

// ---------------- elementwise / norm / scan kernels ----------------
__global__ void k_embed(const int* __restrict__ ids, const float* __restrict__ emb,
                        float* __restrict__ x, float* __restrict__ res)
{
  int i = blockIdx.x * blockDim.x + threadIdx.x;
  if (i >= C_L * C_D) return;
  int l = i / C_D, d = i - l * C_D;
  x[i] = emb[(size_t)ids[l] * C_D + d];
  res[i] = 0.f;
}

// norm_out[row] = rmsnorm(a[row]+b[row]) * w ; optionally sum_out[row] = a[row]+b[row]
__global__ __launch_bounds__(256) void k_addnorm(
    const float* __restrict__ a, const float* __restrict__ b,
    const float* __restrict__ w, float* __restrict__ sum_out,
    float* __restrict__ norm_out)
{
  const int row = blockIdx.x, tid = threadIdx.x;
  const float* ar = a + (size_t)row * C_D;
  const float* br = b + (size_t)row * C_D;
  float v[3]; float ssq = 0.f;
#pragma unroll
  for (int i = 0; i < 3; ++i) {
    int d = tid + i * 256;
    float s = ar[d] + br[d];
    v[i] = s; ssq += s * s;
  }
  __shared__ float red[256];
  red[tid] = ssq; __syncthreads();
  for (int o = 128; o > 0; o >>= 1) { if (tid < o) red[tid] += red[tid + o]; __syncthreads(); }
  const float inv = rsqrtf(red[0] * (1.f / C_D) + C_EPS);
#pragma unroll
  for (int i = 0; i < 3; ++i) {
    int d = tid + i * 256;
    if (sum_out) sum_out[(size_t)row * C_D + d] = v[i];
    norm_out[(size_t)row * C_D + d] = v[i] * inv * w[d];
  }
}

// causal depthwise conv (K=4) + SiLU over xs = xz[:, :DI]
__global__ void k_conv_silu(const float* __restrict__ xz, const float* __restrict__ cw,
                            const float* __restrict__ cb, float* __restrict__ u)
{
  int i = blockIdx.x * blockDim.x + threadIdx.x;
  if (i >= C_L * C_DI) return;
  int l = i / C_DI, d = i - l * C_DI;
  float acc = cb[d];
#pragma unroll
  for (int k = 0; k < 4; ++k) {
    int ls = l - 3 + k;
    if (ls >= 0) acc += xz[(size_t)ls * (2 * C_DI) + d] * cw[d * 4 + k];
  }
  u[i] = acc / (1.f + __expf(-acc));
}

// selective scan: one thread per (channel d, state n); 16-lane shuffle reduction
// (wave32 holds two channels). Sequential over L; gate by silu(z).
__global__ __launch_bounds__(256) void k_scan(
    const float* __restrict__ dbl, const float* __restrict__ dt,
    const float* __restrict__ u, const float* __restrict__ xz,
    const float* __restrict__ Alog, const float* __restrict__ Dv,
    float* __restrict__ y)
{
  const int i = blockIdx.x * blockDim.x + threadIdx.x;   // DI*16 threads
  const int d = i >> 4;
  const int n = i & 15;
  const float a  = -__expf(Alog[(size_t)d * C_N + n]);
  const float Dd = Dv[d];
  float st = 0.f;
  for (int l = 0; l < C_L; ++l) {
    const float dtl = dt[(size_t)l * C_DI + d];
    const float ul  = u[(size_t)l * C_DI + d];
    const float* row = dbl + (size_t)l * C_DBL;
    st = st * __expf(dtl * a) + dtl * row[C_RK + n] * ul;
    float c = st * row[C_RK + C_N + n];
    c += __shfl_xor(c, 1);
    c += __shfl_xor(c, 2);
    c += __shfl_xor(c, 4);
    c += __shfl_xor(c, 8);
    if (n == 0) {
      const float z = xz[(size_t)l * (2 * C_DI) + C_DI + d];
      float yv = c + ul * Dd;
      y[(size_t)l * C_DI + d] = yv * z / (1.f + __expf(-z));
    }
  }
}

__global__ void k_addrow(float* __restrict__ x, const float* __restrict__ v)
{
  int i = blockIdx.x * blockDim.x + threadIdx.x;
  if (i >= C_L * C_D) return;
  x[i] += v[i % C_D];
}

__global__ void k_copy(const float* __restrict__ s, float* __restrict__ d, int n)
{
  int i = blockIdx.x * blockDim.x + threadIdx.x;
  if (i < n) d[i] = s[i];
}

// last-token rmsnorm broadcast into a 16-row staging tile (so WMMA GEMM can use M=16)
__global__ __launch_bounds__(256) void k_normlast(
    const float* __restrict__ x, const float* __restrict__ res,
    const float* __restrict__ w, float* __restrict__ out16)
{
  const int tid = threadIdx.x, row = C_L - 1;
  float v[3]; float ssq = 0.f;
#pragma unroll
  for (int i = 0; i < 3; ++i) {
    int d = tid + i * 256;
    float s = x[(size_t)row * C_D + d] + res[(size_t)row * C_D + d];
    v[i] = s; ssq += s * s;
  }
  __shared__ float red[256];
  red[tid] = ssq; __syncthreads();
  for (int o = 128; o > 0; o >>= 1) { if (tid < o) red[tid] += red[tid + o]; __syncthreads(); }
  const float inv = rsqrtf(red[0] * (1.f / C_D) + C_EPS);
#pragma unroll
  for (int i = 0; i < 3; ++i) {
    int d = tid + i * 256;
    float g = v[i] * inv * w[d];
    for (int r = 0; r < 16; ++r) out16[(size_t)r * C_D + d] = g;
  }
}

// p_max = 1 / sum(exp(logit - max));  flag = (p_max > THR)
__global__ __launch_bounds__(256) void k_conf(const float* __restrict__ lg,
                                              float* __restrict__ flag)
{
  const int tid = threadIdx.x;
  __shared__ float red[256];
  float mx = -3.4e38f;
  for (int v = tid; v < C_V; v += 256) mx = fmaxf(mx, lg[v]);
  red[tid] = mx; __syncthreads();
  for (int o = 128; o > 0; o >>= 1) { if (tid < o) red[tid] = fmaxf(red[tid], red[tid + o]); __syncthreads(); }
  const float m = red[0];
  __syncthreads();
  float se = 0.f;
  for (int v = tid; v < C_V; v += 256) se += __expf(lg[v] - m);
  red[tid] = se; __syncthreads();
  for (int o = 128; o > 0; o >>= 1) { if (tid < o) red[tid] += red[tid + o]; __syncthreads(); }
  if (tid == 0) flag[0] = (1.f / red[0] > C_THR) ? 1.f : 0.f;
}

// select (snapshot if early-exit flag) and final rmsnorm for logits; write loops_taken
__global__ __launch_bounds__(256) void k_normsel(
    const float* __restrict__ x, const float* __restrict__ res,
    const float* __restrict__ xs, const float* __restrict__ rs,
    const float* __restrict__ w, const float* __restrict__ flag,
    float* __restrict__ g, float* __restrict__ loops_out)
{
  const int row = blockIdx.x, tid = threadIdx.x;
  const bool snap = flag[0] > 0.5f;
  const float* A = snap ? xs : x;
  const float* B = snap ? rs : res;
  float v[3]; float ssq = 0.f;
#pragma unroll
  for (int i = 0; i < 3; ++i) {
    int d = tid + i * 256;
    float s = A[(size_t)row * C_D + d] + B[(size_t)row * C_D + d];
    v[i] = s; ssq += s * s;
  }
  __shared__ float red[256];
  red[tid] = ssq; __syncthreads();
  for (int o = 128; o > 0; o >>= 1) { if (tid < o) red[tid] += red[tid + o]; __syncthreads(); }
  const float inv = rsqrtf(red[0] * (1.f / C_D) + C_EPS);
#pragma unroll
  for (int i = 0; i < 3; ++i) {
    int d = tid + i * 256;
    g[(size_t)row * C_D + d] = v[i] * inv * w[d];
  }
  if (row == 0 && tid == 0) loops_out[0] = snap ? 1.f : 2.f;
}

// ---------------- host side ----------------
static inline void gemm(hipStream_t s, const float* act, int lda, const float* W,
                        const float* bias, float* C, int ldc,
                        int M, int N, int K, float alpha, int accum, int epi)
{
  const int tn = (N + 15) / 16;
  if (M % 64 == 0) {
    int tiles = (M / 64) * tn;
    int blocks = (tiles + 7) / 8;       // 8 waves per block
    k_gemm<4><<<blocks, 256, 0, s>>>(act, lda, W, bias, C, ldc, M, N, K, alpha, accum, epi);
  } else {
    int tiles = (M / 16) * tn;
    int blocks = (tiles + 7) / 8;
    k_gemm<1><<<blocks, 256, 0, s>>>(act, lda, W, bias, C, ldc, M, N, K, alpha, accum, epi);
  }
}

struct Bufs { float *x, *res, *h, *xz, *u, *dt, *y, *dbl, *tmp8; };

static void run_mixer(hipStream_t s, const Bufs& B,
                      const float* nw, const float* inw, const float* cw, const float* cb,
                      const float* xw, const float* dtw, const float* dtb,
                      const float* Al, const float* Dp, const float* ow,
                      const float* iA, const float* iB, const float* xA, const float* xB,
                      const float* dA, const float* dB)
{
  // res += x ; h = rmsnorm(res)
  k_addnorm<<<C_L, 256, 0, s>>>(B.x, B.res, nw, B.res, B.h);
  // xz = h @ in_w^T (+ LoRA)
  gemm(s, B.h, C_D, inw, nullptr, B.xz, 2 * C_DI, C_L, 2 * C_DI, C_D, 1.f, 0, EPI_NONE);
  if (iA) {
    gemm(s, B.h, C_D, iA, nullptr, B.tmp8, C_LR, C_L, C_LR, C_D, 1.f, 0, EPI_NONE);
    gemm(s, B.tmp8, C_LR, iB, nullptr, B.xz, 2 * C_DI, C_L, 2 * C_DI, C_LR, C_LS, 1, EPI_NONE);
  }
  k_conv_silu<<<(C_L * C_DI) / 256, 256, 0, s>>>(B.xz, cw, cb, B.u);
  // dbl = u @ x_w^T (+ LoRA)
  gemm(s, B.u, C_DI, xw, nullptr, B.dbl, C_DBL, C_L, C_DBL, C_DI, 1.f, 0, EPI_NONE);
  if (xA) {
    gemm(s, B.u, C_DI, xA, nullptr, B.tmp8, C_LR, C_L, C_LR, C_DI, 1.f, 0, EPI_NONE);
    gemm(s, B.tmp8, C_LR, xB, nullptr, B.dbl, C_DBL, C_L, C_DBL, C_LR, C_LS, 1, EPI_NONE);
  }
  // dt = softplus(dbl[:, :48] @ dt_w^T (+ LoRA) + dt_b)
  if (dA) {
    gemm(s, B.dbl, C_DBL, dA, nullptr, B.tmp8, C_LR, C_L, C_LR, C_RK, 1.f, 0, EPI_NONE);
    gemm(s, B.dbl, C_DBL, dtw, nullptr, B.dt, C_DI, C_L, C_DI, C_RK, 1.f, 0, EPI_NONE);
    gemm(s, B.tmp8, C_LR, dB, dtb, B.dt, C_DI, C_L, C_DI, C_LR, C_LS, 1, EPI_BIAS_SOFTPLUS);
  } else {
    gemm(s, B.dbl, C_DBL, dtw, dtb, B.dt, C_DI, C_L, C_DI, C_RK, 1.f, 0, EPI_BIAS_SOFTPLUS);
  }
  // SSM scan + gate
  k_scan<<<(C_DI * C_N) / 256, 256, 0, s>>>(B.dbl, B.dt, B.u, B.xz, Al, Dp, B.y);
  // x = y @ out_w^T
  gemm(s, B.y, C_DI, ow, nullptr, B.x, C_D, C_L, C_D, C_DI, 1.f, 0, EPI_NONE);
}

extern "C" void kernel_launch(void* const* d_in, const int* in_sizes, int n_in,
                              void* d_out, int out_size, void* d_ws, size_t ws_size,
                              hipStream_t stream)
{
  (void)in_sizes; (void)n_in; (void)ws_size;
  const int*   ids    = (const int*)  d_in[0];
  const float* emb    = (const float*)d_in[1];
  const float* b_norm = (const float*)d_in[2];
  const float* b_inw  = (const float*)d_in[3];
  const float* b_cw   = (const float*)d_in[4];
  const float* b_cb   = (const float*)d_in[5];
  const float* b_xw   = (const float*)d_in[6];
  const float* b_dtw  = (const float*)d_in[7];
  const float* b_dtb  = (const float*)d_in[8];
  const float* b_Al   = (const float*)d_in[9];
  const float* b_Dp   = (const float*)d_in[10];
  const float* b_ow   = (const float*)d_in[11];
  const float* t_norm = (const float*)d_in[12];
  const float* t_inw  = (const float*)d_in[13];
  const float* t_cw   = (const float*)d_in[14];
  const float* t_cb   = (const float*)d_in[15];
  const float* t_xw   = (const float*)d_in[16];
  const float* t_dtw  = (const float*)d_in[17];
  const float* t_dtb  = (const float*)d_in[18];
  const float* t_Al   = (const float*)d_in[19];
  const float* t_Dp   = (const float*)d_in[20];
  const float* t_ow   = (const float*)d_in[21];
  const float* laiA   = (const float*)d_in[22];
  const float* laiB   = (const float*)d_in[23];
  const float* laxA   = (const float*)d_in[24];
  const float* laxB   = (const float*)d_in[25];
  const float* ladA   = (const float*)d_in[26];
  const float* ladB   = (const float*)d_in[27];
  const float* stepw  = (const float*)d_in[28];
  const float* loopw  = (const float*)d_in[29];
  const float* normf  = (const float*)d_in[30];

  // workspace layout (~17 MB of f32)
  float* w = (float*)d_ws;
  float* x     = w; w += C_L * C_D;
  float* res   = w; w += C_L * C_D;
  float* h     = w; w += C_L * C_D;
  float* base  = w; w += C_L * C_D;
  float* xsnap = w; w += C_L * C_D;
  float* rsnap = w; w += C_L * C_D;
  float* g     = w; w += C_L * C_D;
  float* xz    = w; w += C_L * 2 * C_DI;
  float* u     = w; w += C_L * C_DI;
  float* dt    = w; w += C_L * C_DI;
  float* y     = w; w += C_L * C_DI;
  float* dbl   = w; w += C_L * C_DBL;
  float* tmp8  = w; w += C_L * C_LR;
  float* g16   = w; w += 16 * C_D;
  float* lg16  = w; w += 16 * C_V;
  float* flag  = w; w += 16;

  Bufs B{ x, res, h, xz, u, dt, y, dbl, tmp8 };

  // embedding + res=0
  k_embed<<<(C_L * C_D) / 256, 256, 0, stream>>>(ids, emb, x, res);

  // base layers
  for (int i = 0; i < C_NB; ++i) {
    run_mixer(stream, B,
              b_norm + (size_t)i * C_D,
              b_inw  + (size_t)i * 2 * C_DI * C_D,
              b_cw   + (size_t)i * C_DI * 4,
              b_cb   + (size_t)i * C_DI,
              b_xw   + (size_t)i * C_DBL * C_DI,
              b_dtw  + (size_t)i * C_DI * C_RK,
              b_dtb  + (size_t)i * C_DI,
              b_Al   + (size_t)i * C_DI * C_N,
              b_Dp   + (size_t)i * C_DI,
              b_ow   + (size_t)i * C_D * C_DI,
              nullptr, nullptr, nullptr, nullptr, nullptr, nullptr);
  }
  k_copy<<<(C_L * C_D + 255) / 256, 256, 0, stream>>>(x, base, C_L * C_D);

  // recursive loops (host-unrolled; both always run, result selected by device flag)
  for (int step = 0; step < 2; ++step) {
    k_addrow<<<(C_L * C_D) / 256, 256, 0, stream>>>(x, stepw + (size_t)step * C_D);
    for (int j = 0; j < C_NT; ++j) {
      run_mixer(stream, B,
                t_norm + (size_t)j * C_D,
                t_inw  + (size_t)j * 2 * C_DI * C_D,
                t_cw   + (size_t)j * C_DI * 4,
                t_cb   + (size_t)j * C_DI,
                t_xw   + (size_t)j * C_DBL * C_DI,
                t_dtw  + (size_t)j * C_DI * C_RK,
                t_dtb  + (size_t)j * C_DI,
                t_Al   + (size_t)j * C_DI * C_N,
                t_Dp   + (size_t)j * C_DI,
                t_ow   + (size_t)j * C_D * C_DI,
                laiA + (size_t)j * C_LR * C_D,
                laiB + (size_t)j * 2 * C_DI * C_LR,
                laxA + (size_t)j * C_LR * C_DI,
                laxB + (size_t)j * C_DBL * C_LR,
                ladA + (size_t)j * C_LR * C_RK,
                ladB + (size_t)j * C_DI * C_LR);
    }
    // x = rmsnorm(x + base, loop_norm_w)
    k_addnorm<<<C_L, 256, 0, stream>>>(x, base, loopw, nullptr, x);
    if (step == 0) {
      // confidence: last-token logits over the full vocab, softmax max
      k_normlast<<<1, 256, 0, stream>>>(x, res, normf, g16);
      gemm(stream, g16, C_D, emb, nullptr, lg16, C_V, 16, C_V, C_D, 1.f, 0, EPI_NONE);
      k_conf<<<1, 256, 0, stream>>>(lg16, flag);
      // snapshot loop-1 state for the early-exit path
      k_copy<<<(C_L * C_D + 255) / 256, 256, 0, stream>>>(x, xsnap, C_L * C_D);
      k_copy<<<(C_L * C_D + 255) / 256, 256, 0, stream>>>(res, rsnap, C_L * C_D);
    }
  }

  // final: g = rmsnorm(sel(x)+sel(res), normf); logits = g @ emb^T
  float* loops_out = (out_size > C_L * C_V) ? ((float*)d_out + (size_t)C_L * C_V)
                                            : (flag + 1);
  k_normsel<<<C_L, 256, 0, stream>>>(x, res, xsnap, rsnap, normf, flag, g, loops_out);
  gemm(stream, g, C_D, emb, nullptr, (float*)d_out, C_V, C_L, C_V, C_D, 1.f, 0, EPI_NONE);
}
